// CountingAbstraction_80977313398937
// MI455X (gfx1250) — matrix operand order repaired and checked by reading, not compiled, not run on hardware
//
#include <hip/hip_runtime.h>
#include <hip/hip_bf16.h>
#include <math.h>

// ---------------- types / helpers ----------------
typedef __attribute__((ext_vector_type(16))) unsigned short v16u;
typedef __attribute__((ext_vector_type(8)))  unsigned short v8u;
typedef __attribute__((ext_vector_type(16))) __bf16        v16bf;
typedef __attribute__((ext_vector_type(8)))  float         v8f;

#define BB 4
#define NN 4096
#define DD 512
#define MM 512
#define KP 544   // padded width of [rowsum | v_out] (17*32)

__device__ __forceinline__ unsigned short f2bf(float f) {
    unsigned int u = __float_as_uint(f);
    u += 0x7FFFu + ((u >> 16) & 1u);           // round-to-nearest-even
    return (unsigned short)(u >> 16);
}

// A-fragment (16x32 bf16, M x K): lane lo -> row M=lo.
// halves 0..7  <- p[0..7]   (K = hiSel*8 + w)
// halves 8..15 <- p[16..23] (K = 16 + hiSel*8 + w)
// caller passes p = rowbase + k0 + hiSel*8
__device__ __forceinline__ v16bf load_afrag(const unsigned short* p) {
    v8u a = *(const v8u*)(p);
    v8u b = *(const v8u*)(p + 16);
    v16u r;
#pragma unroll
    for (int i = 0; i < 8; ++i) { r[i] = a[i]; r[i + 8] = b[i]; }
    return __builtin_bit_cast(v16bf, r);
}

// B-fragment (32x16 bf16, K x N): lane lo -> col N=lo.
// halves 0..7  <- p[0..7]   (K = hiSel*8 + w)
// halves 8..15 <- p[16..23] (K = 16 + hiSel*8 + w)
// caller passes p = colbase + k0 + hiSel*8
__device__ __forceinline__ v16bf load_bfrag(const unsigned short* p) {
    v8u a = *(const v8u*)(p);
    v8u b = *(const v8u*)(p + 16);
    v16u r;
#pragma unroll
    for (int i = 0; i < 8; ++i) { r[i] = a[i]; r[i + 8] = b[i]; }
    return __builtin_bit_cast(v16bf, r);
}

__device__ __forceinline__ v8f wmma_bf16(v16bf a, v16bf b, v8f c) {
    return __builtin_amdgcn_wmma_f32_16x16x32_bf16(false, a, false, b, (short)0, c, false, false);
}

// ---------------- stage A: l2 normalize + bf16 casts ----------------
__global__ __launch_bounds__(256) void k_normalize(const float* __restrict__ data,
                                                   unsigned short* __restrict__ cn,
                                                   unsigned short* __restrict__ d16) {
    int row = blockIdx.x;                       // over B*N
    int tid = threadIdx.x;
    const float* p = data + (size_t)row * DD;
    float s = 0.f;
    for (int i = tid; i < DD; i += 256) { float v = p[i]; s += v * v; }
    __shared__ float sh[256];
    sh[tid] = s; __syncthreads();
    for (int off = 128; off > 0; off >>= 1) {
        if (tid < off) sh[tid] += sh[tid + off];
        __syncthreads();
    }
    float inv = rsqrtf(fmaxf(sh[0], 1e-12f));
    for (int i = tid; i < DD; i += 256) {
        float v = p[i];
        cn [(size_t)row * DD + i] = f2bf(v * inv);
        d16[(size_t)row * DD + i] = f2bf(v);
    }
}

// ---------------- stage B: V' transposed: [KP][N] bf16 ----------------
// feat 0 = ones (rowsum column), feats 1..512 = posenc, 513..543 = 0
__global__ __launch_bounds__(256) void k_posenc(unsigned short* __restrict__ vpT) {
    int idx = blockIdx.x * 256 + threadIdx.x;   // over KP*N
    int f = idx / NN, n = idx % NN;
    float val;
    if (f == 0)       val = 1.0f;
    else if (f <= DD) {
        int d = f - 1;
        float ang = (float)n * powf(10000.0f, -(float)(d >> 1) / 256.0f);
        val = (d & 1) ? cosf(ang) : sinf(ang);
    } else            val = 0.0f;
    vpT[idx] = f2bf(val);
}

// ---------------- pack weights transposed to bf16 ----------------
__global__ __launch_bounds__(256) void k_pack_exp(const float* __restrict__ W,  // [513,512]
                                                  unsigned short* __restrict__ wT) { // [512][544]
    int idx = blockIdx.x * 256 + threadIdx.x;   // over 512*544
    int n = idx / KP, k = idx % KP;
    wT[idx] = (k < DD + 1) ? f2bf(W[(size_t)k * MM + n]) : (unsigned short)0;
}
__global__ __launch_bounds__(256) void k_pack_merge(const float* __restrict__ W, // [1024,512]
                                                    unsigned short* __restrict__ wT) { // [512][1024]
    int idx = blockIdx.x * 256 + threadIdx.x;   // over 512*1024
    int n = idx / 1024, k = idx % 1024;
    wT[idx] = f2bf(W[(size_t)k * MM + n]);
}

// ---------------- stage C: fused sim = relu(cn cn^T); cpre = sim @ V' ----------------
// block: 16 query rows, 8 waves. Per iter: 128 keys.
//  - query rows (16x512 bf16 = 16KB) staged once in LDS; A-fragments come from ds_load
//  - wave w computes S subtile (16 rows x 16 keys), relu, stage in LDS (bf16)
//  - all waves consume S [16x128] as A-fragments and accumulate O[16 x 544]
__global__ __launch_bounds__(256) void k_simfuse(const unsigned short* __restrict__ cn,
                                                 const unsigned short* __restrict__ vpT,
                                                 unsigned short* __restrict__ cpre) {
    int b     = blockIdx.x >> 8;                // N/16 = 256 row tiles per batch
    int i0    = (blockIdx.x & 255) << 4;
    int wave  = threadIdx.x >> 5;
    int lane  = threadIdx.x & 31;
    int lo    = lane & 15;
    int hiSel = lane >> 4;

    __shared__ __align__(32) unsigned short sA[16 * DD];   // query rows, 16 KB
    __shared__ __align__(32) unsigned short sS[16 * 128];  // S staging, 4 KB

    const unsigned short* cnb = cn + (size_t)b * NN * DD;

    // ---- stage the 16 query rows into LDS (one contiguous 16 KB copy) ----
    {
        const unsigned short* src = cnb + (size_t)i0 * DD;
#pragma unroll
        for (int c = 0; c < 4; ++c) {
            int e = (threadIdx.x + c * 256) * 8;           // v8u = 16 bytes
            *(v8u*)&sA[e] = *(const v8u*)&src[e];
        }
    }
    __syncthreads();

    v8f acc[5];
#pragma unroll
    for (int i = 0; i < 5; ++i) acc[i] = (v8f){};

    for (int j0 = 0; j0 < NN; j0 += 128) {
        // ---- 1) S subtile for keys [j0 + wave*16, +16) ----
        v8f c = (v8f){};
        const unsigned short* pb0 = cnb + (size_t)(j0 + wave * 16 + lo) * DD + hiSel * 8;
        const unsigned short* pa0 = &sA[lo * DD + hiSel * 8];
#pragma unroll 4
        for (int kt = 0; kt < 16; ++kt) {
            v16bf a  = load_afrag(pa0 + kt * 32);          // ds_load
            v16bf bm = load_bfrag(pb0 + kt * 32);          // global_load
            c = wmma_bf16(a, bm, c);
        }
        __syncthreads();                         // prior readers of sS done
#pragma unroll
        for (int r = 0; r < 8; ++r) {
            float v = fmaxf(c[r], 0.0f);         // relu
            sS[(r + 8 * hiSel) * 128 + wave * 16 + lo] = f2bf(v);
        }
        __syncthreads();                         // sS fully written

        // ---- 2) O += S[16x128] @ V'[128 x 544] ----
#pragma unroll
        for (int c4 = 0; c4 < 4; ++c4) {
            v16bf sa = load_afrag(&sS[lo * 128 + c4 * 32 + hiSel * 8]);
#pragma unroll
            for (int idx = 0; idx < 5; ++idx) {
                int t = wave + idx * 8;
                if (t < 34) {
                    int n0 = t * 16;
                    const unsigned short* pv =
                        vpT + (size_t)(n0 + lo) * NN + j0 + c4 * 32 + hiSel * 8;
                    acc[idx] = wmma_bf16(sa, load_bfrag(pv), acc[idx]);
                }
            }
        }
    }

    // ---- store counter-pre (bf16, [B*N][544]) ----
#pragma unroll
    for (int idx = 0; idx < 5; ++idx) {
        int t = wave + idx * 8;
        if (t < 34) {
            int n0 = t * 16;
#pragma unroll
            for (int r = 0; r < 8; ++r) {
                int row = i0 + r + 8 * hiSel;
                cpre[((size_t)b * NN + row) * KP + n0 + lo] = f2bf(acc[idx][r]);
            }
        }
    }
}

// ---------------- stage D: counter = softplus(cpre @ W_exp + b) ----------------
__global__ __launch_bounds__(256) void k_expand(const unsigned short* __restrict__ cpre,
                                                const unsigned short* __restrict__ wexpT,
                                                const float* __restrict__ bexp,
                                                unsigned short* __restrict__ counter) {
    int row0  = blockIdx.x << 4;                 // over B*N rows
    int wave  = threadIdx.x >> 5;
    int lane  = threadIdx.x & 31;
    int lo    = lane & 15;
    int hiSel = lane >> 4;

    v8f acc[4];
#pragma unroll
    for (int i = 0; i < 4; ++i) acc[i] = (v8f){};

    const unsigned short* pa0 = cpre + (size_t)(row0 + lo) * KP + hiSel * 8;
    for (int kt = 0; kt < KP; kt += 32) {
        v16bf a = load_afrag(pa0 + kt);
#pragma unroll
        for (int t = 0; t < 4; ++t) {
            int n0 = (wave * 4 + t) * 16;
            const unsigned short* pb = wexpT + (size_t)(n0 + lo) * KP + kt + hiSel * 8;
            acc[t] = wmma_bf16(a, load_bfrag(pb), acc[t]);
        }
    }
#pragma unroll
    for (int t = 0; t < 4; ++t) {
        int n0 = (wave * 4 + t) * 16;
        float bias = bexp[n0 + lo];
#pragma unroll
        for (int r = 0; r < 8; ++r) {
            float x  = acc[t][r] + bias;
            float sp = fmaxf(x, 0.0f) + log1pf(expf(-fabsf(x)));   // stable softplus
            counter[(size_t)(row0 + r + 8 * hiSel) * MM + n0 + lo] = f2bf(sp);
        }
    }
}

// ---------------- stage E: out = [data | counter] @ W_merge ----------------
__global__ __launch_bounds__(256) void k_merge(const unsigned short* __restrict__ d16,
                                               const unsigned short* __restrict__ counter,
                                               const unsigned short* __restrict__ wmergeT,
                                               float* __restrict__ out) {
    int row0  = blockIdx.x << 4;                 // over B*N rows
    int wave  = threadIdx.x >> 5;
    int lane  = threadIdx.x & 31;
    int lo    = lane & 15;
    int hiSel = lane >> 4;

    v8f acc[4];
#pragma unroll
    for (int i = 0; i < 4; ++i) acc[i] = (v8f){};

    for (int kt = 0; kt < 1024; kt += 32) {
        const unsigned short* src = (kt < DD) ? d16 : counter;
        int koff = (kt < DD) ? kt : (kt - DD);
        v16bf a = load_afrag(src + (size_t)(row0 + lo) * DD + koff + hiSel * 8);
#pragma unroll
        for (int t = 0; t < 4; ++t) {
            int n0 = (wave * 4 + t) * 16;
            const unsigned short* pb = wmergeT + (size_t)(n0 + lo) * 1024 + kt + hiSel * 8;
            acc[t] = wmma_bf16(a, load_bfrag(pb), acc[t]);
        }
    }
#pragma unroll
    for (int t = 0; t < 4; ++t) {
        int n0 = (wave * 4 + t) * 16;
#pragma unroll
        for (int r = 0; r < 8; ++r)
            out[(size_t)(row0 + r + 8 * hiSel) * MM + n0 + lo] = acc[t][r];
    }
}

// ---------------- launcher ----------------
extern "C" void kernel_launch(void* const* d_in, const int* in_sizes, int n_in,
                              void* d_out, int out_size, void* d_ws, size_t ws_size,
                              hipStream_t stream) {
    const float* data    = (const float*)d_in[0];   // [4,4096,512]
    const float* W_exp   = (const float*)d_in[1];   // [513,512]
    const float* b_exp   = (const float*)d_in[2];   // [512]
    const float* W_merge = (const float*)d_in[3];   // [1024,512]
    float* out = (float*)d_out;                     // [4,4096,512] fp32

    char* ws = (char*)d_ws;
    unsigned short* cn      = (unsigned short*)ws;  ws += (size_t)BB * NN * DD * 2;   // 16 MB
    unsigned short* d16     = (unsigned short*)ws;  ws += (size_t)BB * NN * DD * 2;   // 16 MB
    unsigned short* vpT     = (unsigned short*)ws;  ws += (size_t)KP * NN * 2;        // 4.45 MB
    unsigned short* wexpT   = (unsigned short*)ws;  ws += (size_t)MM * KP * 2;        // 0.56 MB
    unsigned short* wmergeT = (unsigned short*)ws;  ws += (size_t)MM * 1024 * 2;      // 1 MB
    unsigned short* cpre    = (unsigned short*)ws;  ws += (size_t)BB * NN * KP * 2;   // 17.8 MB
    unsigned short* counter = (unsigned short*)ws;  ws += (size_t)BB * NN * MM * 2;   // 16 MB

    k_normalize <<<BB * NN,            256, 0, stream>>>(data, cn, d16);
    k_posenc    <<<(KP * NN) / 256,    256, 0, stream>>>(vpT);
    k_pack_exp  <<<(MM * KP) / 256,    256, 0, stream>>>(W_exp, wexpT);
    k_pack_merge<<<(MM * 1024) / 256,  256, 0, stream>>>(W_merge, wmergeT);
    k_simfuse   <<<BB * (NN / 16),     256, 0, stream>>>(cn, vpT, cpre);
    k_expand    <<<(BB * NN) / 16,     256, 0, stream>>>(cpre, wexpT, b_exp, counter);
    k_merge     <<<(BB * NN) / 16,     256, 0, stream>>>(d16, counter, wmergeT, out);
}
